// Experts_8297876815944
// MI455X (gfx1250) — matrix operand order repaired and checked by reading, not compiled, not run on hardware
//
#include <hip/hip_runtime.h>
#include <math.h>

#define DIM      1024
#define D_STATE  16
#define D_CONV   4
#define D_INNER  2048
#define DT_RANK  64
#define B_SZ     2
#define L_SEQ    1024
#define ROWS     (B_SZ * L_SEQ)              /* 2048 */
#define XDBL_W   (DT_RANK + 2 * D_STATE)     /* 96   */

typedef __bf16 v8bf  __attribute__((ext_vector_type(8)));
typedef __bf16 v16bf __attribute__((ext_vector_type(16)));
typedef float  v8f   __attribute__((ext_vector_type(8)));

__device__ __forceinline__ float sigmoidf_(float x) { return 1.0f / (1.0f + __expf(-x)); }
__device__ __forceinline__ float softplusf_(float x) {
    return fmaxf(x, 0.0f) + __logf(1.0f + __expf(-fabsf(x)));
}

// A-fragment (16x32 bf16): elems 0..7 at p, elems 8..15 at p+16
__device__ __forceinline__ v16bf load_afrag(const __bf16* p) {
    const v8bf lo = *(const v8bf*)p;
    const v8bf hi = *(const v8bf*)(p + 16);
    return __builtin_shufflevector(lo, hi, 0,1,2,3,4,5,6,7,8,9,10,11,12,13,14,15);
}

// -----------------------------------------------------------------------------
// f32 -> bf16 bulk convert
// -----------------------------------------------------------------------------
__global__ void cvt_bf16_kernel(const float* __restrict__ in,
                                __bf16* __restrict__ out, int n)
{
    const int i = (blockIdx.x * blockDim.x + threadIdx.x) * 4;
    if (i + 3 < n) {
        const float4 v = *(const float4*)(in + i);
        out[i + 0] = (__bf16)v.x;
        out[i + 1] = (__bf16)v.y;
        out[i + 2] = (__bf16)v.z;
        out[i + 3] = (__bf16)v.w;
    } else {
        for (int j = i; j < n; ++j) out[j] = (__bf16)in[j];
    }
}

// -----------------------------------------------------------------------------
// C[M,N] = A[M,K] x W[N,K]^T, bf16 operands, f32 accum (v_wmma_f32_16x16x32_bf16).
// Wave owns (MT*16)x(NT*16). K-loop is manually unrolled 2x with explicit
// ping-pong fragment buffers: compute on buf0 while loading buf1 and vice
// versa -> zero register copies, loads overlap WMMAs. Requires K % 64 == 0
// (true for all call sites). __launch_bounds__(...,1) = full VGPR budget.
// mode 0: plain store.  mode 1: softplus(acc + bias[col]).
// -----------------------------------------------------------------------------
template<int MT, int NT>
__global__ void __launch_bounds__(256, 1)
gemm_bf16_wmma(const __bf16* __restrict__ A, int lda,
               const __bf16* __restrict__ W, int ldw,
               float* __restrict__ C, int ldc,
               __bf16* __restrict__ Cbf,
               int M, int N, int K,
               const float* __restrict__ bias, int mode)
{
    const int wave    = blockIdx.x * (blockDim.x >> 5) + (threadIdx.x >> 5);
    const int tilesN  = N >> 4;
    const int tilesM  = M >> 4;
    const int groupsN = tilesN / NT;
    const int groupsM = tilesM / MT;
    if (wave >= groupsM * groupsN) return;           // wave-uniform
    const int tm0  = (wave / groupsN) * MT;
    const int tn0  = (wave % groupsN) * NT;
    const int lane = threadIdx.x & 31;
    const int half = lane >> 4;
    const int mrow = lane & 15;

    // per-lane 32-bit element offsets; bases stay uniform (saddr addressing)
    int aoff[MT], woff[NT];
#pragma unroll
    for (int i = 0; i < MT; ++i)
        aoff[i] = ((tm0 + i) * 16 + mrow) * lda + half * 8;
#pragma unroll
    for (int j = 0; j < NT; ++j)
        woff[j] = ((tn0 + j) * 16 + mrow) * ldw + half * 8;

    v8f acc[MT][NT] = {};
    v16bf af0[MT], wf0[NT], af1[MT], wf1[NT];

    auto load0 = [&](int k) {
#pragma unroll
        for (int i = 0; i < MT; ++i) af0[i] = load_afrag(A + aoff[i] + k);
#pragma unroll
        for (int j = 0; j < NT; ++j) wf0[j] = load_afrag(W + woff[j] + k);
    };
    auto load1 = [&](int k) {
#pragma unroll
        for (int i = 0; i < MT; ++i) af1[i] = load_afrag(A + aoff[i] + k);
#pragma unroll
        for (int j = 0; j < NT; ++j) wf1[j] = load_afrag(W + woff[j] + k);
    };
    auto mma0 = [&]() {
#pragma unroll
        for (int i = 0; i < MT; ++i)
#pragma unroll
            for (int j = 0; j < NT; ++j)
                acc[i][j] = __builtin_amdgcn_wmma_f32_16x16x32_bf16(
                    false, af0[i], false, wf0[j], (short)0, acc[i][j], false, false);
    };
    auto mma1 = [&]() {
#pragma unroll
        for (int i = 0; i < MT; ++i)
#pragma unroll
            for (int j = 0; j < NT; ++j)
                acc[i][j] = __builtin_amdgcn_wmma_f32_16x16x32_bf16(
                    false, af1[i], false, wf1[j], (short)0, acc[i][j], false, false);
    };

    load0(0);
    int k0 = 0;
    for (; k0 + 64 < K; k0 += 64) {     // K % 64 == 0 at every call site
        load1(k0 + 32);                 // overlap with mma0
        mma0();
        load0(k0 + 64);                 // overlap with mma1
        mma1();
    }
    // tail: exactly 64 K-elements left, buf0 already holds k0
    load1(k0 + 32);
    mma0();
    mma1();

    // epilogue: C/D layout -> M = r + 8*half, N = mrow
#pragma unroll
    for (int j = 0; j < NT; ++j) {
        const int col = (tn0 + j) * 16 + mrow;
        const float b = (mode == 1) ? bias[col] : 0.0f;
#pragma unroll
        for (int i = 0; i < MT; ++i) {
#pragma unroll
            for (int r = 0; r < 8; ++r) {
                const int row = (tm0 + i) * 16 + r + 8 * half;
                float v = acc[i][j][r];
                if (mode == 1) v = softplusf_(v + b);
                C[(size_t)row * ldc + col] = v;
                if (Cbf) Cbf[(size_t)row * ldc + col] = (__bf16)v;
            }
        }
    }
}

// -----------------------------------------------------------------------------
// Causal depthwise conv1d (width 4) + bias + SiLU; f32 + fused bf16 shadow.
// -----------------------------------------------------------------------------
__global__ void conv_silu_kernel(const float* __restrict__ xz,
                                 const float* __restrict__ conv_w,
                                 const float* __restrict__ conv_b,
                                 float* __restrict__ xin,
                                 __bf16* __restrict__ xin_bf)
{
    const int idx = blockIdx.x * blockDim.x + threadIdx.x;
    if (idx >= ROWS * D_INNER) return;
    const int d  = idx % D_INNER;
    const int gl = idx / D_INNER;
    const int b  = gl / L_SEQ;
    const int l  = gl % L_SEQ;

    float acc = conv_b[d];
#pragma unroll
    for (int k = 0; k < D_CONV; ++k) {
        const int lk = l + k - (D_CONV - 1);
        if (lk >= 0)
            acc += conv_w[d * D_CONV + k] *
                   xz[(size_t)(b * L_SEQ + lk) * (2 * D_INNER) + d];
    }
    const float v = acc * sigmoidf_(acc);
    xin   [(size_t)gl * D_INNER + d] = v;
    xin_bf[(size_t)gl * D_INNER + d] = (__bf16)v;
}

// -----------------------------------------------------------------------------
// Selective scan + skip + SiLU gate, fused; y emitted as bf16 for out_proj.
// gid&15 = state n, gid>>4 = channel; 16-lane shfl_xor butterfly reduction.
// -----------------------------------------------------------------------------
__global__ void selective_scan_kernel(const float* __restrict__ delta,
                                      const float* __restrict__ xdbl,
                                      const float* __restrict__ xin,
                                      const float* __restrict__ A_log,
                                      const float* __restrict__ Dvec,
                                      const float* __restrict__ xz,
                                      __bf16* __restrict__ y_bf)
{
    const int gid = blockIdx.x * blockDim.x + threadIdx.x;   // 0 .. 65535
    const int n   = gid & 15;
    const int ch  = gid >> 4;                                // b*D_INNER + d
    const int b   = ch / D_INNER;
    const int d   = ch % D_INNER;

    const float Acoef = -__expf(A_log[d * D_STATE + n]);
    const float Dd    = Dvec[d];
    float h = 0.0f;
    const size_t rowBase = (size_t)b * L_SEQ;

    for (int l = 0; l < L_SEQ; ++l) {
        const size_t row = rowBase + l;
        const float dt = delta[row * D_INNER + d];
        const float u  = xin  [row * D_INNER + d];
        const float Bn = xdbl [row * XDBL_W + DT_RANK + n];
        const float Cn = xdbl [row * XDBL_W + DT_RANK + D_STATE + n];

        const float dA = __expf(dt * Acoef);
        h = dA * h + dt * Bn * u;

        float part = h * Cn;
        part += __shfl_xor(part, 1, 32);
        part += __shfl_xor(part, 2, 32);
        part += __shfl_xor(part, 4, 32);
        part += __shfl_xor(part, 8, 32);

        if (n == 0) {
            float yv = part + u * Dd;
            const float zv = xz[row * (2 * D_INNER) + D_INNER + d];
            yv *= zv * sigmoidf_(zv);
            y_bf[row * D_INNER + d] = (__bf16)yv;
        }
    }
}

// -----------------------------------------------------------------------------
static inline char* align256(char* p) {
    return (char*)(((uintptr_t)p + 255) & ~(uintptr_t)255);
}

extern "C" void kernel_launch(void* const* d_in, const int* in_sizes, int n_in,
                              void* d_out, int out_size, void* d_ws, size_t ws_size,
                              hipStream_t stream)
{
    const float* x          = (const float*)d_in[0];
    const float* in_proj_w  = (const float*)d_in[1];
    const float* conv_w     = (const float*)d_in[2];
    const float* conv_b     = (const float*)d_in[3];
    const float* x_proj_w   = (const float*)d_in[4];
    const float* dt_proj_w  = (const float*)d_in[5];
    const float* dt_proj_b  = (const float*)d_in[6];
    const float* A_log      = (const float*)d_in[7];
    const float* Dvec       = (const float*)d_in[8];
    const float* out_proj_w = (const float*)d_in[9];
    float* out = (float*)d_out;

    // ---- workspace carve-out (f32 buffers + bf16 shadows), 256B-aligned ----
    char* cur = (char*)d_ws;
    float* xz      = (float*)cur; cur = align256(cur + sizeof(float) * (size_t)ROWS * 2 * D_INNER);
    float* xin     = (float*)cur; cur = align256(cur + sizeof(float) * (size_t)ROWS * D_INNER);
    float* xdbl    = (float*)cur; cur = align256(cur + sizeof(float) * (size_t)ROWS * XDBL_W);
    float* delta   = (float*)cur; cur = align256(cur + sizeof(float) * (size_t)ROWS * D_INNER);
    __bf16* x_bf   = (__bf16*)cur; cur = align256(cur + sizeof(__bf16) * (size_t)ROWS * DIM);
    __bf16* xin_bf = (__bf16*)cur; cur = align256(cur + sizeof(__bf16) * (size_t)ROWS * D_INNER);
    __bf16* xdbl_bf= (__bf16*)cur; cur = align256(cur + sizeof(__bf16) * (size_t)ROWS * XDBL_W);
    __bf16* y_bf   = (__bf16*)cur; cur = align256(cur + sizeof(__bf16) * (size_t)ROWS * D_INNER);
    __bf16* w_in   = (__bf16*)cur; cur = align256(cur + sizeof(__bf16) * (size_t)2 * D_INNER * DIM);
    __bf16* w_xp   = (__bf16*)cur; cur = align256(cur + sizeof(__bf16) * (size_t)XDBL_W * D_INNER);
    __bf16* w_dt   = (__bf16*)cur; cur = align256(cur + sizeof(__bf16) * (size_t)D_INNER * DT_RANK);
    __bf16* w_out  = (__bf16*)cur; cur = align256(cur + sizeof(__bf16) * (size_t)DIM * D_INNER);

    const int BLK = 256, WPB = BLK / 32;
    auto cvt = [&](const float* src, __bf16* dst, size_t n) {
        cvt_bf16_kernel<<<((int)n / 4 + BLK - 1) / BLK, BLK, 0, stream>>>(src, dst, (int)n);
    };

    // 0) bf16 conversion of input + all GEMM weights (pure-bandwidth, ~2us)
    cvt(x,          x_bf,  (size_t)ROWS * DIM);
    cvt(in_proj_w,  w_in,  (size_t)2 * D_INNER * DIM);
    cvt(x_proj_w,   w_xp,  (size_t)XDBL_W * D_INNER);
    cvt(dt_proj_w,  w_dt,  (size_t)D_INNER * DT_RANK);
    cvt(out_proj_w, w_out, (size_t)DIM * D_INNER);

    // 1) xz = x @ in_proj_w^T          (2048 x 4096 x 1024), 32x64 per wave
    {
        const int waves = (ROWS / 32) * ((2 * D_INNER) / 64);
        gemm_bf16_wmma<2, 4><<<(waves + WPB - 1) / WPB, BLK, 0, stream>>>(
            x_bf, DIM, w_in, DIM, xz, 2 * D_INNER, nullptr,
            ROWS, 2 * D_INNER, DIM, nullptr, 0);
    }
    // 2) depthwise conv + SiLU -> xin (f32) + xin_bf (bf16)
    conv_silu_kernel<<<(ROWS * D_INNER + BLK - 1) / BLK, BLK, 0, stream>>>(
        xz, conv_w, conv_b, xin, xin_bf);
    // 3) x_dbl = xin @ x_proj_w^T      (2048 x 96 x 2048), 32x96 per wave
    {
        const int waves = (ROWS / 32) * (XDBL_W / 96);
        gemm_bf16_wmma<2, 6><<<(waves + WPB - 1) / WPB, BLK, 0, stream>>>(
            xin_bf, D_INNER, w_xp, D_INNER, xdbl, XDBL_W, xdbl_bf,
            ROWS, XDBL_W, D_INNER, nullptr, 0);
    }
    // 4) delta = softplus(x_dbl[:, :64] @ dt_proj_w^T + dt_proj_b)
    {
        const int waves = (ROWS / 32) * (D_INNER / 64);
        gemm_bf16_wmma<2, 4><<<(waves + WPB - 1) / WPB, BLK, 0, stream>>>(
            xdbl_bf, XDBL_W, w_dt, DT_RANK, delta, D_INNER, nullptr,
            ROWS, D_INNER, DT_RANK, dt_proj_b, 1);
    }
    // 5) selective scan + skip + gate -> y (bf16)
    selective_scan_kernel<<<(B_SZ * D_INNER * D_STATE) / BLK, BLK, 0, stream>>>(
        delta, xdbl, xin, A_log, Dvec, xz, y_bf);
    // 6) out = y @ out_proj_w^T        (2048 x 1024 x 2048)
    {
        const int waves = (ROWS / 32) * (DIM / 64);
        gemm_bf16_wmma<2, 4><<<(waves + WPB - 1) / WPB, BLK, 0, stream>>>(
            y_bf, D_INNER, w_out, D_INNER, out, DIM, nullptr,
            ROWS, DIM, D_INNER, nullptr, 0);
    }
}
